// InterFrameBiLevelRoutingAttention_22402549416186
// MI455X (gfx1250) — matrix operand order repaired and checked by reading, not compiled, not run on hardware
//
#include <hip/hip_runtime.h>
#include <hip/hip_bf16.h>

typedef __attribute__((ext_vector_type(16))) _Float16 v16h;
typedef __attribute__((ext_vector_type(8)))  _Float16 v8h;
typedef __attribute__((ext_vector_type(8)))  float    v8f;

#define DIM_C    256
#define QK_DIM_C 256
#define KV_C     512   // QK_DIM + DIM
#define QKV_C    768
#define NFRM     16    // N*D
#define NWIN     49    // 7*7
#define W2       64    // 8*8 pixels per window
#define NP       (NFRM*NWIN)          // 784
#define NTOK     (NP*W2)              // 50176
#define NHEAD    8
#define CH       32
#define TOPK_N   4
#define SCALE_F  0.0625f              // 256^-0.5

union AFrag { v16h v; v8h p[2]; };

// A-fragment (16x32 f16, row-major source)
// lane 0-15: row=lane, K range [h*16, h*16+16) with h = lane>>4
__device__ inline v16h load_a(const _Float16* rowptr /* = base + row*lda + h*16 */) {
    AFrag a;
    a.p[0] = *(const v8h*)(rowptr);
    a.p[1] = *(const v8h*)(rowptr + 8);
    return a.v;
}

// ---------------- K1: window-partition + f32->f16 ----------------
__global__ void pack_x(const float* __restrict__ x, _Float16* __restrict__ xh) {
    long t = blockIdx.x;          // token row in window layout
    int  c = threadIdx.x;         // channel
    int pix = t & 63;
    long np = t >> 6;
    int p = (int)(np % NWIN);
    long n = np / NWIN;
    int Ni = (int)(n >> 3), dd = (int)(n & 7);
    int Hi = (p / 7) * 8 + (pix >> 3);
    int Wi = (p % 7) * 8 + (pix & 7);
    long src = ((((long)Ni * 8 + dd) * 56 + Hi) * 56 + Wi) * DIM_C + c;
    xh[t * DIM_C + c] = (_Float16)x[src];
}

// ---------------- K2: transpose weights to [N][K] f16 ----------------
__global__ void pack_w(const float* __restrict__ Wqkv, const float* __restrict__ Wo,
                       _Float16* __restrict__ WqkvT, _Float16* __restrict__ WoT) {
    int idx = blockIdx.x * 256 + threadIdx.x;   // grid 768*256 threads
    if (idx < QKV_C * DIM_C) {
        int o = idx / DIM_C, c = idx % DIM_C;
        WqkvT[idx] = (_Float16)Wqkv[c * QKV_C + o];
    }
    if (idx < DIM_C * DIM_C) {
        int o = idx / DIM_C, c = idx % DIM_C;
        WoT[idx] = (_Float16)Wo[c * DIM_C + o];
    }
}

// ---------------- K3: QKV GEMM (WMMA, 2x2 register-blocked) ----------------
// 32x32 output block per wave: rows 1568 blocks x cols 24 blocks = 37632 wave-jobs
__global__ void qkv_gemm(const _Float16* __restrict__ xh, const _Float16* __restrict__ WqkvT,
                         const float* __restrict__ bqkv,
                         _Float16* __restrict__ qh, _Float16* __restrict__ kvh) {
    int wave = threadIdx.x >> 5;
    int lane = threadIdx.x & 31;
    long job = (long)blockIdx.x * 8 + wave;     // exact: 37632 jobs
    int rb = (int)(job / 24), cb = (int)(job % 24);
    int row0 = rb * 32, col0 = cb * 32;
    int r = lane & 15, h = lane >> 4;

    const _Float16* A0 = xh    + (long)(row0 + r) * DIM_C + h * 16;
    const _Float16* A1 = A0 + 16 * DIM_C;
    const _Float16* B0 = WqkvT + (long)(col0 + r) * DIM_C + h * 16;
    const _Float16* B1 = B0 + 16 * DIM_C;
    __builtin_prefetch(B0, 0, 1);               // global_prefetch_b8 (weight panel)

    v8f acc00 = {}, acc01 = {}, acc10 = {}, acc11 = {};
#pragma unroll
    for (int s = 0; s < 8; ++s) {
        v16h a0 = load_a(A0 + s * 32);
        v16h a1 = load_a(A1 + s * 32);
        v16h b0 = *(const v16h*)(B0 + s * 32);
        v16h b1 = *(const v16h*)(B1 + s * 32);
        acc00 = __builtin_amdgcn_wmma_f32_16x16x32_f16(false, a0, false, b0, (short)0, acc00, false, false);
        acc01 = __builtin_amdgcn_wmma_f32_16x16x32_f16(false, a0, false, b1, (short)0, acc01, false, false);
        acc10 = __builtin_amdgcn_wmma_f32_16x16x32_f16(false, a1, false, b0, (short)0, acc10, false, false);
        acc11 = __builtin_amdgcn_wmma_f32_16x16x32_f16(false, a1, false, b1, (short)0, acc11, false, false);
    }
    float bias0 = bqkv[col0 + r];
    float bias1 = bqkv[col0 + 16 + r];
    if (col0 < QK_DIM_C) {                      // q output, pre-scaled
#pragma unroll
        for (int v = 0; v < 8; ++v) {
            long ra = row0 + v + h * 8;
            long rbr = ra + 16;
            qh[ra  * QK_DIM_C + col0 + r]      = (_Float16)((acc00[v] + bias0) * SCALE_F);
            qh[ra  * QK_DIM_C + col0 + 16 + r] = (_Float16)((acc01[v] + bias1) * SCALE_F);
            qh[rbr * QK_DIM_C + col0 + r]      = (_Float16)((acc10[v] + bias0) * SCALE_F);
            qh[rbr * QK_DIM_C + col0 + 16 + r] = (_Float16)((acc11[v] + bias1) * SCALE_F);
        }
    } else {                                    // kv output (unshifted)
        int c0 = col0 - QK_DIM_C;
#pragma unroll
        for (int v = 0; v < 8; ++v) {
            long ra = row0 + v + h * 8;
            long rbr = ra + 16;
            kvh[ra  * KV_C + c0 + r]      = (_Float16)(acc00[v] + bias0);
            kvh[ra  * KV_C + c0 + 16 + r] = (_Float16)(acc01[v] + bias1);
            kvh[rbr * KV_C + c0 + r]      = (_Float16)(acc10[v] + bias0);
            kvh[rbr * KV_C + c0 + 16 + r] = (_Float16)(acc11[v] + bias1);
        }
    }
}

// ---------------- K4: window means ----------------
__global__ void means_kernel(const _Float16* __restrict__ qh, const _Float16* __restrict__ kvh,
                             float* __restrict__ q_win, float* __restrict__ k_win) {
    int np = blockIdx.x;        // 784
    int c  = threadIdx.x;       // 256
    long rb = (long)np * W2;
    float sq = 0.f, sk = 0.f;
    for (int pix = 0; pix < W2; ++pix) {
        sq += (float)qh[(rb + pix) * QK_DIM_C + c];
        sk += (float)kvh[(rb + pix) * KV_C + c];
    }
    q_win[(long)np * DIM_C + c] = sq * (1.f / 64.f);
    k_win[(long)np * DIM_C + c] = sk * (1.f / 64.f);
}

// ---------------- K5: routing logits + top-4 ----------------
__global__ void route_topk(const float* __restrict__ q_win, const float* __restrict__ k_win,
                           int* __restrict__ r_idx) {
    int np = blockIdx.x;                  // 784
    int n = np / NWIN;
    int d = n & 7;
    int n_src = (d < 7) ? n + 1 : n;      // frame shift
    __shared__ float logit[NWIN];
    int q2 = threadIdx.x;
    if (q2 < NWIN) {
        const float* qv = q_win + (long)np * DIM_C;               // already SCALE'd
        const float* kv = k_win + ((long)n_src * NWIN + q2) * DIM_C;
        float s = 0.f;
        for (int c = 0; c < DIM_C; ++c) s += qv[c] * kv[c];
        logit[q2] = s;
    }
    __syncthreads();
    if (threadIdx.x == 0) {
        for (int t = 0; t < TOPK_N; ++t) {
            float best = -1e30f; int bi = 0;
            for (int q3 = 0; q3 < NWIN; ++q3)
                if (logit[q3] > best) { best = logit[q3]; bi = q3; }
            r_idx[np * TOPK_N + t] = bi;
            logit[bi] = -1e30f;
        }
    }
}

// ---------------- K6: bi-level attention (WMMA) ----------------
// one block per (n, p, head); 64 threads = 2 waves; each wave does 2 M-tiles of 16 queries
__global__ void attn_kernel(const _Float16* __restrict__ qh, const _Float16* __restrict__ kvh,
                            const int* __restrict__ r_idx, _Float16* __restrict__ ao) {
    int m  = blockIdx.x & 7;
    int np = blockIdx.x >> 3;
    int n  = np / NWIN;
    int d  = n & 7;
    int n_src = (d < 7) ? n + 1 : n;
    long qbase = (long)np * W2;                   // query rows for this (n,p)
    long kvb   = (long)n_src * NWIN * W2;         // kv row base for source frame

    __shared__ __align__(16) _Float16 vT[CH][256];       // V transposed: [ch][key]  16KB
    __shared__ __align__(16) _Float16 Pl[2][16][256];    // per-wave P tiles         16KB
    __shared__ int sel[TOPK_N];

    if (threadIdx.x < TOPK_N) sel[threadIdx.x] = r_idx[np * TOPK_N + threadIdx.x];
    __syncthreads();

    // gather V for this head into LDS, transposed
    {
        int key = threadIdx.x;                    // 64 threads -> 4 keys each
        for (int kk = 0; kk < 4; ++kk, key += 64) {
            int selw = key >> 6, pix = key & 63;
            long srow = kvb + (long)sel[selw] * W2 + pix;
            const _Float16* src = kvh + srow * KV_C + QK_DIM_C + m * CH;
#pragma unroll
            for (int g = 0; g < 4; ++g) {
                v8h val = *(const v8h*)(src + g * 8);
#pragma unroll
                for (int j = 0; j < 8; ++j) vT[g * 8 + j][key] = val[j];
            }
        }
    }
    __syncthreads();

    int wave = threadIdx.x >> 5;
    int lane = threadIdx.x & 31;
    int r = lane & 15, h = lane >> 4;

    for (int tt = 0; tt < 2; ++tt) {
        int t = wave * 2 + tt;                    // M-tile 0..3
        // Q fragment (already scaled), K-dim = 32 (one WMMA step)
        const _Float16* Aq = qh + (qbase + t * 16 + r) * QK_DIM_C + m * CH + h * 16;
        v16h aq = load_a(Aq);

        // logits: 16 key-tiles, full 256 keys
        v8f S[16];
#pragma unroll
        for (int kt = 0; kt < 16; ++kt) {
            int key = kt * 16 + r;
            int selw = key >> 6, pix = key & 63;
            long srow = kvb + (long)sel[selw] * W2 + pix;
            v16h b = *(const v16h*)(kvh + srow * KV_C + m * CH + h * 16);
            v8f z = {};
            S[kt] = __builtin_amdgcn_wmma_f32_16x16x32_f16(false, aq, false, b, (short)0, z, false, false);
        }

        // softmax over 256 keys per row (row = v + h*8; cols striped over 16 lanes x 16 tiles)
#pragma unroll
        for (int v = 0; v < 8; ++v) {
            float mx = S[0][v];
#pragma unroll
            for (int kt = 1; kt < 16; ++kt) mx = fmaxf(mx, S[kt][v]);
#pragma unroll
            for (int off = 1; off < 16; off <<= 1) mx = fmaxf(mx, __shfl_xor(mx, off, 32));
            float sum = 0.f;
#pragma unroll
            for (int kt = 0; kt < 16; ++kt) {
                float e = __expf(S[kt][v] - mx);
                S[kt][v] = e;
                sum += e;
            }
#pragma unroll
            for (int off = 1; off < 16; off <<= 1) sum += __shfl_xor(sum, off, 32);
            float inv = 1.f / sum;
#pragma unroll
            for (int kt = 0; kt < 16; ++kt) S[kt][v] *= inv;
        }

        // C-layout -> A-layout via per-wave LDS region (no barrier needed: same wave)
#pragma unroll
        for (int v = 0; v < 8; ++v) {
            int row = v + h * 8;
#pragma unroll
            for (int kt = 0; kt < 16; ++kt)
                Pl[wave][row][kt * 16 + r] = (_Float16)S[kt][v];
        }

        // out = P @ V : K = 256 keys (8 steps), N = 32 ch (2 tiles)
        v8f O0 = {}, O1 = {};
#pragma unroll
        for (int s = 0; s < 8; ++s) {
            v16h ap = load_a(&Pl[wave][r][s * 32 + h * 16]);
            v16h b0 = *(const v16h*)(&vT[r][s * 32 + h * 16]);
            v16h b1 = *(const v16h*)(&vT[16 + r][s * 32 + h * 16]);
            O0 = __builtin_amdgcn_wmma_f32_16x16x32_f16(false, ap, false, b0, (short)0, O0, false, false);
            O1 = __builtin_amdgcn_wmma_f32_16x16x32_f16(false, ap, false, b1, (short)0, O1, false, false);
        }

        // store f16 attention output (window layout) for final GEMM
#pragma unroll
        for (int v = 0; v < 8; ++v) {
            long row = qbase + t * 16 + v + h * 8;
            ao[row * DIM_C + m * CH + r]      = (_Float16)O0[v];
            ao[row * DIM_C + m * CH + 16 + r] = (_Float16)O1[v];
        }
    }
}

// ---------------- K7: output GEMM (WMMA, 2x2 register-blocked) + inverse window permute ----------------
__global__ void out_gemm(const _Float16* __restrict__ ao, const _Float16* __restrict__ WoT,
                         const float* __restrict__ bo, float* __restrict__ out) {
    int wave = threadIdx.x >> 5;
    int lane = threadIdx.x & 31;
    long job = (long)blockIdx.x * 8 + wave;     // exact: 1568*8 = 12544 jobs
    int rb = (int)(job >> 3), cb = (int)(job & 7);
    int row0 = rb * 32, col0 = cb * 32;
    int r = lane & 15, h = lane >> 4;

    const _Float16* A0 = ao  + (long)(row0 + r) * DIM_C + h * 16;
    const _Float16* A1 = A0 + 16 * DIM_C;
    const _Float16* B0 = WoT + (long)(col0 + r) * DIM_C + h * 16;
    const _Float16* B1 = B0 + 16 * DIM_C;
    __builtin_prefetch(B0, 0, 1);               // global_prefetch_b8 (weight panel)

    v8f acc00 = {}, acc01 = {}, acc10 = {}, acc11 = {};
#pragma unroll
    for (int s = 0; s < 8; ++s) {
        v16h a0 = load_a(A0 + s * 32);
        v16h a1 = load_a(A1 + s * 32);
        v16h b0 = *(const v16h*)(B0 + s * 32);
        v16h b1 = *(const v16h*)(B1 + s * 32);
        acc00 = __builtin_amdgcn_wmma_f32_16x16x32_f16(false, a0, false, b0, (short)0, acc00, false, false);
        acc01 = __builtin_amdgcn_wmma_f32_16x16x32_f16(false, a0, false, b1, (short)0, acc01, false, false);
        acc10 = __builtin_amdgcn_wmma_f32_16x16x32_f16(false, a1, false, b0, (short)0, acc10, false, false);
        acc11 = __builtin_amdgcn_wmma_f32_16x16x32_f16(false, a1, false, b1, (short)0, acc11, false, false);
    }
    float bias0 = bo[col0 + r];
    float bias1 = bo[col0 + 16 + r];
#pragma unroll
    for (int i = 0; i < 2; ++i) {
#pragma unroll
        for (int v = 0; v < 8; ++v) {
            long t = row0 + i * 16 + v + h * 8;
            int pix = (int)(t & 63);
            long np = t >> 6;
            int p = (int)(np % NWIN);
            long n = np / NWIN;
            int Ni = (int)(n >> 3), dd = (int)(n & 7);
            int Hi = (p / 7) * 8 + (pix >> 3);
            int Wi = (p % 7) * 8 + (pix & 7);
            long dst = ((((long)Ni * 8 + dd) * 56 + Hi) * 56 + Wi) * DIM_C + col0 + r;
            float v0 = (i == 0 ? acc00[v] : acc10[v]) + bias0;
            float v1 = (i == 0 ? acc01[v] : acc11[v]) + bias1;
            out[dst]      = v0;
            out[dst + 16] = v1;
        }
    }
}

// ---------------- launch ----------------
extern "C" void kernel_launch(void* const* d_in, const int* in_sizes, int n_in,
                              void* d_out, int out_size, void* d_ws, size_t ws_size,
                              hipStream_t stream) {
    const float* x    = (const float*)d_in[0];
    const float* Wqkv = (const float*)d_in[1];
    const float* bqkv = (const float*)d_in[2];
    const float* Wo   = (const float*)d_in[3];
    const float* bo   = (const float*)d_in[4];
    float* out = (float*)d_out;

    // workspace carve-up (bytes), ~130.6 MB total
    char* ws = (char*)d_ws;
    size_t off = 0;
    _Float16* xh     = (_Float16*)(ws + off); off += (size_t)NTOK * DIM_C * 2;        // 25.7MB
    _Float16* WqkvT  = (_Float16*)(ws + off); off += (size_t)QKV_C * DIM_C * 2;       // 384KB
    _Float16* WoT    = (_Float16*)(ws + off); off += (size_t)DIM_C * DIM_C * 2;       // 128KB
    _Float16* qh     = (_Float16*)(ws + off); off += (size_t)NTOK * QK_DIM_C * 2;     // 25.7MB
    _Float16* kvh    = (_Float16*)(ws + off); off += (size_t)NTOK * KV_C * 2;         // 51.4MB
    float*    q_win  = (float*)   (ws + off); off += (size_t)NP * DIM_C * 4;          // 0.8MB
    float*    k_win  = (float*)   (ws + off); off += (size_t)NP * DIM_C * 4;          // 0.8MB
    int*      r_idx  = (int*)     (ws + off); off += (size_t)NP * TOPK_N * 4;         // 12.5KB
    _Float16* ao     = (_Float16*)(ws + off); off += (size_t)NTOK * DIM_C * 2;        // 25.7MB

    pack_x      <<<NTOK,                256, 0, stream>>>(x, xh);
    pack_w      <<<QKV_C,               256, 0, stream>>>(Wqkv, Wo, WqkvT, WoT);
    qkv_gemm    <<<(NTOK/32)*24/8,      256, 0, stream>>>(xh, WqkvT, bqkv, qh, kvh); // 4704 blocks
    means_kernel<<<NP,                  256, 0, stream>>>(qh, kvh, q_win, k_win);
    route_topk  <<<NP,                   64, 0, stream>>>(q_win, k_win, r_idx);
    attn_kernel <<<NP * NHEAD,           64, 0, stream>>>(qh, kvh, r_idx, ao);       // 6272 blocks
    out_gemm    <<<(NTOK/32)*8/8,       256, 0, stream>>>(ao, WoT, bo, out);         // 1568 blocks
}
